// BinaryDense_18339510354746
// MI455X (gfx1250) — 8-bit matrix operand order repaired and checked by reading, compiled, not run on hardware
//
#include <hip/hip_runtime.h>

typedef int v8i __attribute__((ext_vector_type(8)));

#define B_ROWS  8192
#define D_DIM   2048
#define UNITS_N 2048

__global__ __launch_bounds__(256) void pack_signs_rowmajor(
    const float* __restrict__ x, signed char* __restrict__ out) {
  const int base = (blockIdx.x * 256 + threadIdx.x) * 16;
  const float4* xv = (const float4*)(x + base);
  int words[4];
#pragma unroll
  for (int i = 0; i < 4; ++i) {
    float4 f = xv[i];
    int b0 = (f.x >= 0.f) ? 0x01 : 0xFF;
    int b1 = (f.y >= 0.f) ? 0x01 : 0xFF;
    int b2 = (f.z >= 0.f) ? 0x01 : 0xFF;
    int b3 = (f.w >= 0.f) ? 0x01 : 0xFF;
    words[i] = b0 | (b1 << 8) | (b2 << 16) | (b3 << 24);
  }
  int4 r;
  r.x = words[0]; r.y = words[1]; r.z = words[2]; r.w = words[3];
  *(int4*)(out + base) = r;
}

__global__ __launch_bounds__(256) void pack_signs_transpose(
    const float* __restrict__ w, signed char* __restrict__ wt) {
  __shared__ signed char tile[64][68];
  const int k0 = blockIdx.y * 64;
  const int n0 = blockIdx.x * 64;
  const int t  = threadIdx.x;
#pragma unroll
  for (int i = 0; i < 16; ++i) {
    int lin = t + 256 * i;
    int k = lin >> 6;
    int n = lin & 63;
    float f = w[(size_t)(k0 + k) * UNITS_N + (n0 + n)];
    tile[k][n] = (f >= 0.f) ? (signed char)1 : (signed char)-1;
  }
  __syncthreads();
#pragma unroll
  for (int i = 0; i < 4; ++i) {
    int j  = t + 256 * i;
    int n  = j >> 4;
    int kk = (j & 15) * 4;
    int v  = (tile[kk + 0][n] & 0xFF)        |
             ((tile[kk + 1][n] & 0xFF) << 8) |
             ((tile[kk + 2][n] & 0xFF) << 16)|
             ((tile[kk + 3][n] & 0xFF) << 24);
    *(int*)(wt + (size_t)(n0 + n) * D_DIM + (k0 + kk)) = v;
  }
}

__global__ __launch_bounds__(256) void binary_gemm_wmma(
    const signed char* __restrict__ Ap, const signed char* __restrict__ Wt,
    const float* __restrict__ bias, float* __restrict__ out) {
  const int lane = threadIdx.x & 31;
  const int wave = threadIdx.x >> 5;
  const int wm   = wave >> 1;
  const int wn   = wave & 1;
  const int Moff = blockIdx.y * 128 + wm * 32;
  const int Noff = blockIdx.x * 128 + wn * 64;
  const int l16  = lane & 15;
  const int hi   = (lane >> 4) & 1;

  const signed char* aptr[2];
#pragma unroll
  for (int jm = 0; jm < 2; ++jm)
    aptr[jm] = Ap + (size_t)(Moff + jm * 16 + l16) * D_DIM + hi * 16;
  const signed char* bptr[4];
#pragma unroll
  for (int jn = 0; jn < 4; ++jn)
    bptr[jn] = Wt + (size_t)(Noff + jn * 16 + l16) * D_DIM + hi * 16;

  v8i acc[2][4];
#pragma unroll
  for (int jm = 0; jm < 2; ++jm)
#pragma unroll
    for (int jn = 0; jn < 4; ++jn)
      acc[jm][jn] = (v8i){0, 0, 0, 0, 0, 0, 0, 0};

  for (int k0 = 0; k0 < D_DIM; k0 += 64) {
    if (k0 + 64 < D_DIM) {
      __builtin_prefetch(aptr[0] + k0 + 64, 0, 1);
      __builtin_prefetch(bptr[0] + k0 + 64, 0, 1);
    }
    v8i a[2], b[4];
#pragma unroll
    for (int jm = 0; jm < 2; ++jm) {
      const signed char* p = aptr[jm] + k0;
      int4 q0 = *(const int4*)(p + 0);
      int4 q1 = *(const int4*)(p + 32);
      a[jm] = (v8i){q0.x, q0.y, q0.z, q0.w, q1.x, q1.y, q1.z, q1.w};
    }
#pragma unroll
    for (int jn = 0; jn < 4; ++jn) {
      const signed char* p = bptr[jn] + k0;
      int4 u0 = *(const int4*)(p + 0);
      int4 u1 = *(const int4*)(p + 32);
      b[jn] = (v8i){u0.x, u0.y, u0.z, u0.w, u1.x, u1.y, u1.z, u1.w};
    }
#pragma unroll
    for (int jm = 0; jm < 2; ++jm)
#pragma unroll
      for (int jn = 0; jn < 4; ++jn)
        acc[jm][jn] = __builtin_amdgcn_wmma_i32_16x16x64_iu8(
             true, a[jm],  true, b[jn],
            acc[jm][jn],  false,  false);
  }

#pragma unroll
  for (int jn = 0; jn < 4; ++jn) {
    const int   col = Noff + jn * 16 + l16;
    const float bv  = bias[col];
#pragma unroll
    for (int jm = 0; jm < 2; ++jm) {
      const int row0 = Moff + jm * 16 + hi * 8;
      float* o = out + (size_t)row0 * UNITS_N + col;
#pragma unroll
      for (int v = 0; v < 8; ++v)
        o[(size_t)v * UNITS_N] = (float)acc[jm][jn][v] + bv;
    }
  }
}

extern "C" void kernel_launch(void* const* d_in, const int* in_sizes, int n_in,
                              void* d_out, int out_size, void* d_ws, size_t ws_size,
                              hipStream_t stream) {
  (void)in_sizes; (void)n_in; (void)out_size; (void)ws_size;
  const float* x    = (const float*)d_in[0];
  const float* w    = (const float*)d_in[1];
  const float* bias = (const float*)d_in[2];
  float* out = (float*)d_out;

  signed char* Ap = (signed char*)d_ws;
  signed char* Wt = Ap + (size_t)B_ROWS * D_DIM;

  pack_signs_rowmajor<<<(B_ROWS * D_DIM) / (256 * 16), 256, 0, stream>>>(x, Ap);
  pack_signs_transpose<<<dim3(UNITS_N / 64, D_DIM / 64), 256, 0, stream>>>(w, Wt);

  binary_gemm_wmma<<<dim3(UNITS_N / 128, B_ROWS / 128), 256, 0, stream>>>(
      Ap, Wt, bias, out);
}
